// GPEncoder_57818849739248
// MI455X (gfx1250) — matrix operand order repaired and checked by reading, not compiled, not run on hardware
//
#include <hip/hip_runtime.h>
#include <math.h>

typedef __attribute__((ext_vector_type(16))) _Float16 v16h;
typedef __attribute__((ext_vector_type(8)))  float    v8f;

namespace {
constexpr int IMH = 256;
constexpr int IMW = 256;
constexpr int ICH = 3;
constexpr int OCH = 30;
constexpr int KR  = 27;        // real im2col depth (3 ic * 3 * 3)
constexpr int HW  = IMH * IMW;
constexpr float TWO_PI = 6.28318530717958647692f;
}

// One block = one image row (256 px) = 8 wave32; each wave: 32 px.
// Conv via implicit GEMM: A = weights (16 oc x 32 K, f16), B = im2col
// (32 K x 16 px, f16), D = f32 16x16; 2 oc-tiles x 2 px-tiles = 4 WMMA/wave.
__global__ __launch_bounds__(256) void gp_encoder_kernel(
    const float* __restrict__ x,    // (4,3,256,256)
    const float* __restrict__ Wt,   // (30,3,3,3) == (30,27)
    const float* __restrict__ bias, // (30,)
    float* __restrict__ out)        // xf | m | Sout | y (flat, f32)
{
  const int tid    = threadIdx.x;
  const int wv     = tid >> 5;
  const int lane   = tid & 31;
  const int laneHi = lane >> 4;     // 0: lanes 0-15, 1: lanes 16-31
  const int laneLo = lane & 15;

  const int row = blockIdx.x & (IMH - 1);
  const int n   = blockIdx.x >> 8;          // batch index
  const int px0 = wv * 32;                  // first column of this wave

  __shared__ float lds[8][OCH * 32];        // per-wave p staging, 30 KB

  // ---------------- A fragments: weights as f16 ----------------
  // 16-bit A 16x32 layout: lane holds row M = laneLo; half h -> VGPR v=h/2:
  //   v<4: K = 2v + (h&1) + laneHi*8 ; v>=4: K = 16 + 2(v-4) + (h&1) + laneHi*8
  v16h a0, a1;
  #pragma unroll
  for (int h = 0; h < 16; ++h) {
    const int v  = h >> 1, bit = h & 1;
    const int k  = (v < 4 ? 2 * v + bit : 16 + 2 * (v - 4) + bit) + laneHi * 8;
    const int o0 = laneLo;            // oc tile 0: rows 0..15
    const int o1 = 16 + laneLo;       // oc tile 1: rows 16..31 (30,31 pad)
    float w0 = (k < KR) ? Wt[o0 * KR + k] : 0.0f;
    float w1 = (k < KR && o1 < OCH) ? Wt[o1 * KR + k] : 0.0f;
    a0[h] = (_Float16)w0;
    a1[h] = (_Float16)w1;
  }

  // ---------------- conv: 2 pixel-tiles of 16 ----------------
  #pragma unroll
  for (int t = 0; t < 2; ++t) {
    const int col = px0 + t * 16 + laneLo;  // this lane's B column (pixel)
    // 16-bit B 32x16 layout: lane = N; elem h<8: K = h + laneHi*8, h>=8: K = 8 + h + laneHi*8
    v16h bm;
    #pragma unroll
    for (int h = 0; h < 16; ++h) {
      const int k = (h < 8 ? h : h + 8) + laneHi * 8;
      float v = 0.0f;
      if (k < KR) {
        const int ic = k / 9, rem = k % 9;
        const int ky = rem / 3, kx = rem % 3;
        const int yy = row + ky - 1, xx = col + kx - 1;
        if (yy >= 0 && yy < IMH && xx >= 0 && xx < IMW)
          v = x[((n * ICH + ic) * IMH + yy) * IMW + xx];
      }
      bm[h] = (_Float16)v;
    }

    v8f acc0 = {};
    v8f acc1 = {};
    acc0 = __builtin_amdgcn_wmma_f32_16x16x32_f16(false, a0, false, bm,
                                                  (short)0, acc0, false, false);
    acc1 = __builtin_amdgcn_wmma_f32_16x16x32_f16(false, a1, false, bm,
                                                  (short)0, acc1, false, false);

    // D layout: VGPR r, lane -> M = r + laneHi*8, N = laneLo
    #pragma unroll
    for (int r = 0; r < 8; ++r) {
      const int oc = r + laneHi * 8;
      lds[wv][oc * 32 + t * 16 + laneLo] = acc0[r];
      const int oc1 = 16 + oc;
      if (oc1 < OCH) lds[wv][oc1 * 32 + t * 16 + laneLo] = acc1[r];
    }
  }
  __syncthreads();

  // ---------------- per-pixel epilogue: one lane per pixel ----------------
  const int col = px0 + lane;
  float p[OCH];
  #pragma unroll
  for (int c = 0; c < OCH; ++c) p[c] = lds[wv][c * 32 + lane] + bias[c];

  float m6[6], d6[6], id6[6], xf[3];
  #pragma unroll
  for (int c = 0; c < 6; ++c) m6[c] = p[c];
  #pragma unroll
  for (int k = 0; k < 6; ++k) {
    float s = p[6 + k];
    s = s * s;                       // square(relu(s)-relu(-s)) == s^2
    s = fmaxf(s, 1e-6f);             // clip(s, 1e-6)
    d6[k] = s;
    id6[k] = 1.0f / s;
  }
  #pragma unroll
  for (int c = 0; c < 3; ++c) xf[c] = p[27 + c];

  // R = product of 15 Givens rotations applied to identity (column updates)
  float R[6][6];
  #pragma unroll
  for (int i = 0; i < 6; ++i)
    #pragma unroll
    for (int j = 0; j < 6; ++j) R[i][j] = (i == j) ? 1.0f : 0.0f;

  const int PI_[15] = {0,0,0,0,0,1,1,1,1,2,2,2,3,3,4};
  const int PJ_[15] = {1,2,3,4,5,2,3,4,5,3,4,5,4,5,5};
  #pragma unroll
  for (int k = 0; k < 15; ++k) {
    const float t  = p[12 + k];
    const float ak = TWO_PI / (1.0f + expf(-t));   // 2*pi*sigmoid
    const float c  = cosf(ak);
    const float sn = sinf(ak);
    const int i = PI_[k], j = PJ_[k];
    #pragma unroll
    for (int r = 0; r < 6; ++r) {
      const float ci = R[r][i], cj = R[r][j];
      R[r][i] =  c * ci + sn * cj;
      R[r][j] = -sn * ci + c * cj;
    }
  }

  // S = R diag(d) R^T ; S1 = R diag(1/d) R^T
  float S[6][6], S1[6][6];
  #pragma unroll
  for (int mi = 0; mi < 6; ++mi)
    #pragma unroll
    for (int ni = 0; ni < 6; ++ni) {
      float s = 0.0f, s1 = 0.0f;
      #pragma unroll
      for (int k = 0; k < 6; ++k) {
        const float rr = R[mi][k] * R[ni][k];
        s  = fmaf(rr, d6[k],  s);
        s1 = fmaf(rr, id6[k], s1);
      }
      S[mi][ni] = s;
      S1[mi][ni] = s1;
    }

  // inv(Sxx) via adjugate, Sxx = S1[0:3][0:3]
  const float A00 = S1[0][0], A01 = S1[0][1], A02 = S1[0][2];
  const float A10 = S1[1][0], A11 = S1[1][1], A12 = S1[1][2];
  const float A20 = S1[2][0], A21 = S1[2][1], A22 = S1[2][2];
  const float c00 =  (A11 * A22 - A12 * A21);
  const float c01 = -(A10 * A22 - A12 * A20);
  const float c02 =  (A10 * A21 - A11 * A20);
  const float idet = 1.0f / (A00 * c00 + A01 * c01 + A02 * c02);
  float IV[3][3];
  IV[0][0] = c00 * idet;
  IV[0][1] = -(A01 * A22 - A02 * A21) * idet;
  IV[0][2] =  (A01 * A12 - A02 * A11) * idet;
  IV[1][0] = c01 * idet;
  IV[1][1] =  (A00 * A22 - A02 * A20) * idet;
  IV[1][2] = -(A00 * A12 - A02 * A10) * idet;
  IV[2][0] = c02 * idet;
  IV[2][1] = -(A00 * A21 - A01 * A20) * idet;
  IV[2][2] =  (A00 * A11 - A01 * A10) * idet;

  // Q = Syx * inv(Sxx), Syx = S1[3:6][0:3]
  float Q[3][3];
  #pragma unroll
  for (int r = 0; r < 3; ++r)
    #pragma unroll
    for (int c = 0; c < 3; ++c) {
      float q = 0.0f;
      #pragma unroll
      for (int k = 0; k < 3; ++k) q = fmaf(S1[3 + r][k], IV[k][c], q);
      Q[r][c] = q;
    }

  // y = my + Q @ (xf - mx)
  float yv[3];
  #pragma unroll
  for (int r = 0; r < 3; ++r) {
    float acc = m6[3 + r];
    #pragma unroll
    for (int c = 0; c < 3; ++c) acc = fmaf(Q[r][c], xf[c] - m6[c], acc);
    yv[r] = acc;
  }

  // ---------------- stores (layouts per reference output order) ----------
  const int pix = row * IMW + col;
  float* xf_out = out;
  float* m_out  = out + 4 * 3 * HW;
  float* S_out  = out + 4 * 3 * HW + 4 * 6 * HW;
  float* y_out  = out + 4 * 3 * HW + 4 * 6 * HW + 4 * 36 * HW;

  #pragma unroll
  for (int c = 0; c < 3; ++c) xf_out[(n * 3 + c) * HW + pix] = xf[c];
  #pragma unroll
  for (int c = 0; c < 6; ++c) m_out[(n * 6 + c) * HW + pix] = m6[c];
  #pragma unroll
  for (int i = 0; i < 6; ++i)
    #pragma unroll
    for (int j = 0; j < 6; ++j)
      S_out[((n * 6 + i) * 6 + j) * HW + pix] = S[i][j];
  #pragma unroll
  for (int c = 0; c < 3; ++c) y_out[(n * 3 + c) * HW + pix] = yv[c];
}

extern "C" void kernel_launch(void* const* d_in, const int* in_sizes, int n_in,
                              void* d_out, int out_size, void* d_ws, size_t ws_size,
                              hipStream_t stream) {
  (void)in_sizes; (void)n_in; (void)out_size; (void)d_ws; (void)ws_size;
  const float* x  = (const float*)d_in[0];
  const float* Wt = (const float*)d_in[1];
  const float* b  = (const float*)d_in[2];
  float* out = (float*)d_out;

  dim3 grid(4 * IMH);   // batch * rows
  dim3 block(256);      // 8 wave32, one row per block
  hipLaunchKernelGGL(gp_encoder_kernel, grid, block, 0, stream, x, Wt, b, out);
}
